// Attention_65661460021977
// MI455X (gfx1250) — matrix operand order repaired and checked by reading, not compiled, not run on hardware
//
#include <hip/hip_runtime.h>
#include <hip/hip_bf16.h>
#include <math.h>

typedef __attribute__((ext_vector_type(16))) __bf16 bf16x16;
typedef __attribute__((ext_vector_type(8)))  __bf16 bf16x8;
typedef __attribute__((ext_vector_type(4)))  __bf16 bf16x4;
typedef __attribute__((ext_vector_type(8)))  float  f32x8;

constexpr int BM = 128;
constexpr int BN = 128;
constexpr int BK = 32;
constexpr int KP = 40;   // padded K stride (bf16 elems) in LDS: 80B rows, 16B-aligned, bank-friendly

// ---------------------------------------------------------------------------
// f32 -> (hi, lo) bf16 split:  x ~= hi + lo  (~16 effective mantissa bits)
// ---------------------------------------------------------------------------
__global__ void split_f32_to_bf16x2(const float* __restrict__ x,
                                    __bf16* __restrict__ hi,
                                    __bf16* __restrict__ lo, int n) {
    int i = blockIdx.x * blockDim.x + threadIdx.x;
    if (i < n) {
        float v = x[i];
        __bf16 h = (__bf16)v;
        hi[i] = h;
        lo[i] = (__bf16)(v - (float)h);
    }
}

// ---------------------------------------------------------------------------
// Fragment loaders from LDS (layouts per CDNA5 ISA 7.12.2, 16-bit A 16x32 / B 32x16)
// A: lane half lo16 in {0,8};  e<8 -> K=lo16+e,  e>=8 -> K=lo16+8+e  (two b128 reads)
// B: lane half klo in {0,8};  e<8 -> K=klo+e,  e>=8 -> K=klo+8+e, N = lane%16 (two b128 reads)
// LDS tiles are stored K-contiguous: sA[m*KP + k], sB[n*KP + k]
// ---------------------------------------------------------------------------
__device__ __forceinline__ bf16x16 frag_a(const __bf16* s, int m, int lane) {
    int lo16 = (lane < 16) ? 0 : 8;
    const __bf16* p = s + m * KP + lo16;
    bf16x8 a0 = *(const bf16x8*)(p);
    bf16x8 a1 = *(const bf16x8*)(p + 16);
    bf16x16 r;
#pragma unroll
    for (int i = 0; i < 8; ++i) { r[i] = a0[i]; r[i + 8] = a1[i]; }
    return r;
}

__device__ __forceinline__ bf16x16 frag_b(const __bf16* s, int n, int lane) {
    int klo = (lane < 16) ? 0 : 8;
    const __bf16* p = s + n * KP + klo;
    bf16x8 b0 = *(const bf16x8*)(p);
    bf16x8 b1 = *(const bf16x8*)(p + 16);
    bf16x16 r;
#pragma unroll
    for (int i = 0; i < 8; ++i) { r[i] = b0[i]; r[i + 8] = b1[i]; }
    return r;
}

// ---------------------------------------------------------------------------
// C = A*B with bf16x3 split precision: Ah*Bh + Ah*Bl + Al*Bh (f32 accum).
// 128x128 macro tile, BK=32, 256 threads / 8 waves, wave tile = 32x64 (2x4 WMMA tiles).
// Software pipelined: next K-tile is prefetched into registers while the
// current tile is consumed from LDS, then committed to LDS after the barrier.
// B_TRANS:   B(k,n) = Bsrc[n*ldb + k]   (used for Ws^T in pass 2)
// OUT_SPLIT: write C re-split to (Ch, Cl) bf16; else write f32 + bias to Cf.
// ---------------------------------------------------------------------------
template <bool B_TRANS, bool OUT_SPLIT>
__global__ __launch_bounds__(256)
void gemm_bf16x3(const __bf16* __restrict__ Ah, const __bf16* __restrict__ Al, int lda,
                 const __bf16* __restrict__ Bh, const __bf16* __restrict__ Bl, int ldb,
                 int K,
                 __bf16* __restrict__ Ch, __bf16* __restrict__ Cl,
                 float* __restrict__ Cf, int ldc,
                 const float* __restrict__ bias)
{
    __shared__ __bf16 sAh[BM * KP];
    __shared__ __bf16 sAl[BM * KP];
    __shared__ __bf16 sBh[BN * KP];
    __shared__ __bf16 sBl[BN * KP];

    const int tid  = threadIdx.x;
    const int lane = tid & 31;
    const int wave = tid >> 5;
    const int wr   = wave >> 1;   // 0..3 : 32-row strip
    const int wc   = wave & 1;    // 0..1 : 64-col strip
    const int m0   = blockIdx.y * BM;
    const int n0   = blockIdx.x * BN;

    f32x8 acc[2][4];
#pragma unroll
    for (int i = 0; i < 2; ++i)
#pragma unroll
        for (int j = 0; j < 4; ++j)
#pragma unroll
            for (int e = 0; e < 8; ++e) acc[i][j][e] = 0.0f;

    // register staging for the software pipeline (32 VGPRs)
    bf16x4 rah[4], ral[4], rbh[4], rbl[4];

    auto global_fetch = [&](int k0) {
#pragma unroll
        for (int i = 0; i < 4; ++i) {
            int c  = tid + i * 256;
            int r  = c >> 3;            // row in tile (0..127)
            int kc = (c & 7) * 4;       // k offset (0..28)
            size_t g = (size_t)(m0 + r) * lda + (size_t)(k0 + kc);
            rah[i] = *(const bf16x4*)&Ah[g];
            ral[i] = *(const bf16x4*)&Al[g];
        }
        if (B_TRANS) {
#pragma unroll
            for (int i = 0; i < 4; ++i) {
                int c  = tid + i * 256;
                int r  = c >> 3;        // n in tile (0..127)
                int kc = (c & 7) * 4;
                size_t g = (size_t)(n0 + r) * ldb + (size_t)(k0 + kc);
                rbh[i] = *(const bf16x4*)&Bh[g];
                rbl[i] = *(const bf16x4*)&Bl[g];
            }
        } else {
#pragma unroll
            for (int i = 0; i < 4; ++i) {
                int c  = tid + i * 256;
                int kr = c >> 5;        // k in tile (0..31)
                int nc = (c & 31) * 4;  // n in tile (0..124)
                size_t g = (size_t)(k0 + kr) * ldb + (size_t)(n0 + nc);
                rbh[i] = *(const bf16x4*)&Bh[g];
                rbl[i] = *(const bf16x4*)&Bl[g];
            }
        }
    };

    auto lds_commit = [&]() {
#pragma unroll
        for (int i = 0; i < 4; ++i) {
            int c  = tid + i * 256;
            int r  = c >> 3;
            int kc = (c & 7) * 4;
            *(bf16x4*)&sAh[r * KP + kc] = rah[i];
            *(bf16x4*)&sAl[r * KP + kc] = ral[i];
        }
        if (B_TRANS) {
#pragma unroll
            for (int i = 0; i < 4; ++i) {
                int c  = tid + i * 256;
                int r  = c >> 3;
                int kc = (c & 7) * 4;
                *(bf16x4*)&sBh[r * KP + kc] = rbh[i];
                *(bf16x4*)&sBl[r * KP + kc] = rbl[i];
            }
        } else {
#pragma unroll
            for (int i = 0; i < 4; ++i) {
                int c  = tid + i * 256;
                int kr = c >> 5;
                int nc = (c & 31) * 4;
#pragma unroll
                for (int j = 0; j < 4; ++j) {
                    sBh[(nc + j) * KP + kr] = rbh[i][j];
                    sBl[(nc + j) * KP + kr] = rbl[i][j];
                }
            }
        }
    };

    auto compute_step = [&]() {
        bf16x16 ahf[2], alf[2];
#pragma unroll
        for (int mt = 0; mt < 2; ++mt) {
            int m = wr * 32 + mt * 16 + (lane & 15);
            ahf[mt] = frag_a(sAh, m, lane);
            alf[mt] = frag_a(sAl, m, lane);
        }
#pragma unroll
        for (int nt = 0; nt < 4; ++nt) {
            int n = wc * 64 + nt * 16 + (lane & 15);
            bf16x16 bhf = frag_b(sBh, n, lane);
            bf16x16 blf = frag_b(sBl, n, lane);
#pragma unroll
            for (int mt = 0; mt < 2; ++mt) {
                acc[mt][nt] = __builtin_amdgcn_wmma_f32_16x16x32_bf16(
                    false, ahf[mt], false, bhf, (short)0, acc[mt][nt], false, false);
                acc[mt][nt] = __builtin_amdgcn_wmma_f32_16x16x32_bf16(
                    false, ahf[mt], false, blf, (short)0, acc[mt][nt], false, false);
                acc[mt][nt] = __builtin_amdgcn_wmma_f32_16x16x32_bf16(
                    false, alf[mt], false, bhf, (short)0, acc[mt][nt], false, false);
            }
        }
    };

    // ---- software pipeline ----
    global_fetch(0);
    lds_commit();
    __syncthreads();
    for (int k0 = BK; k0 < K; k0 += BK) {
        global_fetch(k0);   // issue next tile's global loads (latency hidden by WMMAs)
        compute_step();     // consume current LDS tile
        __syncthreads();    // all waves done reading LDS
        lds_commit();       // commit prefetched registers to LDS
        __syncthreads();
    }
    compute_step();         // final tile

    // ---- epilogue: C/D layout = VGPR r -> M = (lane<16?0:8)+r, N = lane%16 ----
    float bv = 0.0f;
    if (!OUT_SPLIT && bias) bv = bias[0];
#pragma unroll
    for (int mt = 0; mt < 2; ++mt)
#pragma unroll
        for (int nt = 0; nt < 4; ++nt) {
            int n  = n0 + wc * 64 + nt * 16 + (lane & 15);
            int mb = m0 + wr * 32 + mt * 16 + ((lane < 16) ? 0 : 8);
#pragma unroll
            for (int r = 0; r < 8; ++r) {
                size_t idx = (size_t)(mb + r) * ldc + (size_t)n;
                float v = acc[mt][nt][r];
                if (OUT_SPLIT) {
                    __bf16 h = (__bf16)v;
                    Ch[idx] = h;
                    Cl[idx] = (__bf16)(v - (float)h);
                } else {
                    Cf[idx] = v + bv;
                }
            }
        }
}

// ---------------------------------------------------------------------------
// Row softmax, one block (256 threads) per row of 4096 f32.
// ---------------------------------------------------------------------------
__global__ __launch_bounds__(256)
void softmax_rows(float* __restrict__ A, int ncols) {
    __shared__ float red[256];
    const int tid = threadIdx.x;
    float* p = A + (size_t)blockIdx.x * ncols;
    const int per = ncols / 256;   // 16 for ncols=4096

    float vals[16];
    float mx = -3.4e38f;
#pragma unroll 16
    for (int i = 0; i < per; ++i) {
        vals[i] = p[tid + i * 256];
        mx = fmaxf(mx, vals[i]);
    }
    red[tid] = mx;
    __syncthreads();
    for (int s = 128; s > 0; s >>= 1) {
        if (tid < s) red[tid] = fmaxf(red[tid], red[tid + s]);
        __syncthreads();
    }
    mx = red[0];
    __syncthreads();

    float sum = 0.0f;
#pragma unroll 16
    for (int i = 0; i < per; ++i) {
        vals[i] = __expf(vals[i] - mx);
        sum += vals[i];
    }
    red[tid] = sum;
    __syncthreads();
    for (int s = 128; s > 0; s >>= 1) {
        if (tid < s) red[tid] += red[tid + s];
        __syncthreads();
    }
    float inv = 1.0f / red[0];
#pragma unroll 16
    for (int i = 0; i < per; ++i)
        p[tid + i * 256] = vals[i] * inv;
}

// ---------------------------------------------------------------------------
// Host-side launch
// ---------------------------------------------------------------------------
extern "C" void kernel_launch(void* const* d_in, const int* in_sizes, int n_in,
                              void* d_out, int out_size, void* d_ws, size_t ws_size,
                              hipStream_t stream) {
    const float* Ws   = (const float*)d_in[0];   // [N, D] f32
    const float* W    = (const float*)d_in[1];   // [1, D, D] f32
    const float* bias = (const float*)d_in[2];   // [1] f32

    const int D = (int)(sqrt((double)in_sizes[1]) + 0.5);  // 2048
    const int N = in_sizes[0] / D;                         // 4096

    // Workspace layout (bf16): Wsh|Wsl|Wh|Wl|Th|Tl  (~80 MB)
    __bf16* Wsh = (__bf16*)d_ws;
    __bf16* Wsl = Wsh + (size_t)N * D;
    __bf16* Wh  = Wsl + (size_t)N * D;
    __bf16* Wl  = Wh  + (size_t)D * D;
    __bf16* Th  = Wl  + (size_t)D * D;
    __bf16* Tl  = Th  + (size_t)N * D;

    const int nWs = N * D;
    const int nW  = D * D;
    split_f32_to_bf16x2<<<(nWs + 255) / 256, 256, 0, stream>>>(Ws, Wsh, Wsl, nWs);
    split_f32_to_bf16x2<<<(nW  + 255) / 256, 256, 0, stream>>>(W,  Wh,  Wl,  nW);

    // Pass 1: T[N,D] = Ws @ W   (A = Ws, B = W row-major [k][n])
    dim3 g1(D / BN, N / BM);
    gemm_bf16x3<false, true><<<g1, 256, 0, stream>>>(
        Wsh, Wsl, D, Wh, Wl, D, D, Th, Tl, nullptr, D, nullptr);

    // Pass 2: Aout[N,N] = T @ Ws^T + b   (B(k,n) = Ws[n][k])
    dim3 g2(N / BN, N / BM);
    gemm_bf16x3<true, false><<<g2, 256, 0, stream>>>(
        Th, Tl, D, Wsh, Wsl, D, D, nullptr, nullptr, (float*)d_out, N, bias);

    // Row softmax in place on d_out
    softmax_rows<<<N, 256, 0, stream>>>((float*)d_out, N);
}
